// LocalFeatureExtractor_7559142441576
// MI455X (gfx1250) — matrix operand order repaired and checked by reading, not compiled, not run on hardware
//
#include <hip/hip_runtime.h>
#include <hip/hip_bf16.h>
#include <math.h>

typedef _Float16 v16h __attribute__((ext_vector_type(16)));
typedef _Float16 v8h  __attribute__((ext_vector_type(8)));
typedef _Float16 v4h  __attribute__((ext_vector_type(4)));
typedef _Float16 h2   __attribute__((ext_vector_type(2)));
typedef float    v8f  __attribute__((ext_vector_type(8)));

#define N_PTS 8192
#define DIM   128
#define KNN   16
#define PITCH 264   // f16 elems per LDS row: 256 + 8 pad (breaks 64-bank conflicts)

#define W1T_OFF 0
#define W2T_OFF (256 * PITCH)
#define A_OFF   (W2T_OFF + 128 * PITCH)
#define A_WAVE  (16 * PITCH)
#define LDS_HALFS (A_OFF + 8 * A_WAVE)
#define LDS_BYTES (LDS_HALFS * 2)      // 270336 <= 320KB WGP LDS

// ---------------------------------------------------------------------------
// Kernel 1: brute-force 16-NN over 2-D points. One wave32 per query.
// ---------------------------------------------------------------------------
__global__ __launch_bounds__(256) void knn_kernel(const float* __restrict__ points,
                                                  int* __restrict__ idx_out) {
  __shared__ float    sD[8 * 32 * KNN];
  __shared__ unsigned sI[8 * 32 * KNN];
  const int lane = threadIdx.x & 31;
  const int wave = threadIdx.x >> 5;
  const int q = blockIdx.x * 8 + wave;
  const float2* __restrict__ p2 = (const float2*)points;
  const float2 qp = p2[q];
  const float INF = __uint_as_float(0x7f800000u);

  float    bd[KNN];
  unsigned bi[KNN];
#pragma unroll
  for (int s = 0; s < KNN; ++s) { bd[s] = INF; bi[s] = 0xffffffffu; }

  for (int j = lane; j < N_PTS; j += 32) {
    float2 pj = p2[j];
    float dx = pj.x - qp.x, dy = pj.y - qp.y;
    float d = fmaf(dx, dx, dy * dy);
    d = (j == q) ? INF : d;                // exclude self (reference drops it)
    if (d < bd[KNN - 1]) {                 // wave skips only if ALL lanes reject
      float cd = d; unsigned ci = (unsigned)j;
#pragma unroll
      for (int s = 0; s < KNN; ++s) {
        bool lt = cd < bd[s];
        float td = lt ? bd[s] : cd;  unsigned ti = lt ? bi[s] : ci;
        bd[s] = lt ? cd : bd[s];     bi[s] = lt ? ci : bi[s];
        cd = td; ci = ti;
      }
    }
  }

  const int off = (wave * 32 + lane) * KNN;
#pragma unroll
  for (int s = 0; s < KNN; ++s) { sD[off + s] = bd[s]; sI[off + s] = bi[s]; }
  // DS ops from the same wave are in-order: region is wave-private, no barrier

  int p = 0;
  int myidx = 0;
  const int woff = wave * 32 * KNN;
  for (int r = 0; r < KNN; ++r) {
    float    dv = (p < KNN) ? sD[woff + lane * KNN + p] : __uint_as_float(0x7f800000u);
    unsigned iv = (p < KNN) ? sI[woff + lane * KNN + p] : 0xffffffffu;
    unsigned long long key = ((unsigned long long)__float_as_uint(dv) << 32) | iv;
    unsigned long long m = key;
#pragma unroll
    for (int s = 1; s < 32; s <<= 1) {
      unsigned lo = (unsigned)m, hi = (unsigned)(m >> 32);
      lo = __shfl_xor(lo, s, 32);
      hi = __shfl_xor(hi, s, 32);
      unsigned long long o = ((unsigned long long)hi << 32) | lo;
      m = (o < m) ? o : m;
    }
    if (key == m) p++;                     // unique keys -> exactly one lane advances
    if (lane == r) myidx = (int)(unsigned)(m & 0xffffffffu);
  }
  if (lane < KNN) idx_out[q * KNN + lane] = myidx;
}

// ---------------------------------------------------------------------------
// Kernel 2: fused gather + 2-layer GELU MLP + mean over K, via WMMA f16.
// ---------------------------------------------------------------------------
union AF { v16h v; v8h h[2]; };

// Branch-free exact-GELU: erf via Abramowitz-Stegun 7.1.26 (|err|<=1.5e-7).
__device__ __forceinline__ float gelu_exact(float v) {
  float x  = v * 0.70710678118654752f;     // v / sqrt(2)
  float ax = __builtin_fabsf(x);
  float t  = __builtin_amdgcn_rcpf(fmaf(0.3275911f, ax, 1.0f));
  float p  = fmaf(1.061405429f, t, -1.453152027f);
  p = fmaf(p, t, 1.421413741f);
  p = fmaf(p, t, -0.284496736f);
  p = fmaf(p, t, 0.254829592f);
  p *= t;
  float e    = __builtin_amdgcn_exp2f(ax * ax * -1.4426950408889634f); // exp(-x^2)
  float erfa = fmaf(-p, e, 1.0f);
  float er   = __builtin_copysignf(erfa, x);
  return 0.5f * v * (1.0f + er);
}

__device__ __forceinline__ void load_bfrags(AF* buf, const _Float16* wrow) {
#pragma unroll
  for (int kc = 0; kc < 8; ++kc) {
    buf[kc].h[0] = *(const v8h*)&wrow[32 * kc];
    buf[kc].h[1] = *(const v8h*)&wrow[32 * kc + 16];
  }
}

__device__ __forceinline__ v8f wmma_chain(const AF* a, const AF* b) {
  v8f c = {};
#pragma unroll
  for (int kc = 0; kc < 8; ++kc)
    c = __builtin_amdgcn_wmma_f32_16x16x32_f16(false, a[kc].v, false, b[kc].v,
                                               (short)0, c, false, false);
  return c;
}

__global__ __launch_bounds__(256, 1) void mlp_kernel(const float* __restrict__ x,
                                                     const int* __restrict__ idx,
                                                     const float* __restrict__ W1,
                                                     const float* __restrict__ b1,
                                                     const float* __restrict__ W2,
                                                     const float* __restrict__ b2,
                                                     float* __restrict__ out) {
  extern __shared__ _Float16 lds[];
  const int tid  = threadIdx.x;
  const int lane = tid & 31;
  const int wave = tid >> 5;
  const bool hiHalf = (lane >= 16);
  const int l15 = lane & 15;

  // ---- stage W1^T (256x256) and W2^T (128x256), n-major f16, into LDS ----
  {
    const int n = tid;                       // 0..255
    for (int k = 0; k < 256; k += 2) {
      float f0 = W1[k * 256 + n];
      float f1 = W1[(k + 1) * 256 + n];
      h2 hv = { (_Float16)f0, (_Float16)f1 };
      *(h2*)&lds[W1T_OFF + n * PITCH + k] = hv;
    }
    const int n2 = tid & 127;
    const int kh = (tid >> 7) * 128;
    for (int kk = 0; kk < 128; kk += 2) {
      int k = kh + kk;
      float f0 = W2[k * 128 + n2];
      float f1 = W2[(k + 1) * 128 + n2];
      h2 hv = { (_Float16)f0, (_Float16)f1 };
      *(h2*)&lds[W2T_OFF + n2 * PITCH + k] = hv;
    }
  }
  __syncthreads();

  const int q = blockIdx.x * 8 + wave;
  _Float16* __restrict__ ldsA = &lds[A_OFF + wave * A_WAVE];

  // ---- gather: local row j = [x[idx_j] - x[q], x[q]] -> f16, 16x256 panel ----
  const float4 cv = *(const float4*)&x[q * DIM + 4 * lane];
  v4h ch = { (_Float16)cv.x, (_Float16)cv.y, (_Float16)cv.z, (_Float16)cv.w };
#pragma unroll 4
  for (int j = 0; j < KNN; ++j) {
    int nb = idx[q * KNN + j];
    float4 kv = *(const float4*)&x[nb * DIM + 4 * lane];
    v4h dh = { (_Float16)(kv.x - cv.x), (_Float16)(kv.y - cv.y),
               (_Float16)(kv.z - cv.z), (_Float16)(kv.w - cv.w) };
    *(v4h*)&ldsA[j * PITCH + 4 * lane]       = dh;
    *(v4h*)&ldsA[j * PITCH + 128 + 4 * lane] = ch;
  }

  // ---- GEMM1 A fragments (ISA 16-bit A layout: lane<16 K{0..7,16..23},
  //      lane>=16 K{8..15,24..31} -> two 16B DS loads per 16x32 chunk) ----
  const int m    = l15;
  const int aSel = hiHalf ? 8 : 0;           // A K-offset per lane half
  const int bSel = hiHalf ? 8 : 0;           // B K-offset per lane half
  AF a[8];
#pragma unroll
  for (int kc = 0; kc < 8; ++kc) {
    int k0 = 32 * kc + aSel;
    a[kc].h[0] = *(const v8h*)&ldsA[m * PITCH + k0];
    a[kc].h[1] = *(const v8h*)&ldsA[m * PITCH + k0 + 16];
  }

  // H1 = gelu(local @ W1 + b1): 16 N-tiles, K=256 in 8 chunks of 32.
  // Cross-tile double-buffered B: while the 8-WMMA chain consumes one buffer,
  // the 16 ds_load_b128 for the next tile are in flight.
  {
    const _Float16* w1r = &lds[W1T_OFF + l15 * PITCH + bSel];
    AF bA[8], bB[8];
    load_bfrags(bA, w1r);                    // tile 0
#pragma unroll 1
    for (int tt = 0; tt < 8; ++tt) {
      const int t0 = 2 * tt, t1 = 2 * tt + 1;
      float bias0 = b1[16 * t0 + l15];
      float bias1 = b1[16 * t1 + l15];
      load_bfrags(bB, w1r + (16 * t1) * PITCH);
      {
        v8f c = wmma_chain(a, bA);
#pragma unroll
        for (int r = 0; r < 8; ++r) {
          float g = gelu_exact(c[r] + bias0);
          ldsA[(r + aSel) * PITCH + 16 * t0 + l15] = (_Float16)g;
        }
      }
      load_bfrags(bA, w1r + (16 * ((t1 + 1) & 15)) * PITCH);
      {
        v8f c = wmma_chain(a, bB);
#pragma unroll
        for (int r = 0; r < 8; ++r) {
          float g = gelu_exact(c[r] + bias1);
          ldsA[(r + aSel) * PITCH + 16 * t1 + l15] = (_Float16)g;
        }
      }
    }
  }

  // ---- GEMM2: A2 fragments from H1 panel (in-order DS guarantees H1 visible) ----
  AF a2[8];
#pragma unroll
  for (int kc = 0; kc < 8; ++kc) {
    int k0 = 32 * kc + aSel;
    a2[kc].h[0] = *(const v8h*)&ldsA[m * PITCH + k0];
    a2[kc].h[1] = *(const v8h*)&ldsA[m * PITCH + k0 + 16];
  }

  const float inv16 = 1.0f / 16.0f;
  {
    const _Float16* w2r = &lds[W2T_OFF + l15 * PITCH + bSel];
    AF bA[8], bB[8];
    load_bfrags(bA, w2r);                    // tile 0
#pragma unroll 1
    for (int tt = 0; tt < 4; ++tt) {
      const int t0 = 2 * tt, t1 = 2 * tt + 1;
      float bias0 = b2[16 * t0 + l15];
      float bias1 = b2[16 * t1 + l15];
      load_bfrags(bB, w2r + (16 * t1) * PITCH);
      {
        v8f c = wmma_chain(a2, bA);
        float s = 0.f;
#pragma unroll
        for (int r = 0; r < 8; ++r) s += gelu_exact(c[r] + bias0);
        s += __shfl_xor(s, 16, 32);
        if (!hiHalf) out[q * DIM + 16 * t0 + l15] = s * inv16;
      }
      load_bfrags(bA, w2r + (16 * ((t1 + 1) & 7)) * PITCH);
      {
        v8f c = wmma_chain(a2, bB);
        float s = 0.f;
#pragma unroll
        for (int r = 0; r < 8; ++r) s += gelu_exact(c[r] + bias1);
        s += __shfl_xor(s, 16, 32);
        if (!hiHalf) out[q * DIM + 16 * t1 + l15] = s * inv16;
      }
    }
  }
}

// ---------------------------------------------------------------------------
extern "C" void kernel_launch(void* const* d_in, const int* in_sizes, int n_in,
                              void* d_out, int out_size, void* d_ws, size_t ws_size,
                              hipStream_t stream) {
  const float* x      = (const float*)d_in[0];
  const float* points = (const float*)d_in[1];
  const float* W1     = (const float*)d_in[2];
  const float* b1     = (const float*)d_in[3];
  const float* W2     = (const float*)d_in[4];
  const float* b2     = (const float*)d_in[5];
  float* out = (float*)d_out;
  int* idx = (int*)d_ws;                   // 8192*16 ints = 512 KB scratch

  knn_kernel<<<N_PTS / 8, 256, 0, stream>>>(points, idx);

  hipFuncSetAttribute((const void*)mlp_kernel,
                      hipFuncAttributeMaxDynamicSharedMemorySize, LDS_BYTES);
  mlp_kernel<<<N_PTS / 8, 256, LDS_BYTES, stream>>>(x, idx, W1, b1, W2, b2, out);

  (void)in_sizes; (void)n_in; (void)out_size; (void)ws_size;
}
